// LossEngine_59820304499393
// MI455X (gfx1250) — matrix operand order repaired and checked by reading, not compiled, not run on hardware
//
#include <hip/hip_runtime.h>
#include <math.h>

// CDNA5 / gfx1250, wave32. Heavy math via v_wmma_f32_16x16x32_bf16.
// Register budgets sized to avoid scratch spills:
//   conv:    32x32 C tile, 4 accumulators, fragments loaded per k-step.
//   quant:   16x32 C tile, 2 accumulators, 8 A-frags (64 VGPR) hoisted.
//   ce:      16x32 C tile, 2 accumulators, A streamed (K=768).
// Workspace requirement: ~174 MB.

typedef __attribute__((ext_vector_type(16))) __bf16 v16bf;
typedef __attribute__((ext_vector_type(8)))  __bf16 v8bf;
typedef __attribute__((ext_vector_type(8)))  float  v8f;

#define Q_    8
#define B_    8
#define DIN   512
#define T0_   4096
#define T1_   2048
#define T2_   1024
#define CBS   1024
#define CBD   256
#define HID_  768
#define NROWS (Q_ * B_ * T2_)   // 65536 (q,b,t) rows
#define NPART (NROWS / 16)      // 4096 CE partials

// ---------------- WMMA fragment helpers (wave32 layouts, ISA 7.12.2) --------

// A (16x32 bf16, M x K): lane L -> row M=L%16; elems 0..7 at K=8*half+i,
// elems 8..15 at K=16+8*half+i.  p = row base (+k0).
static __device__ inline v16bf afrag(const __bf16* __restrict__ p, int half) {
  v8bf lo = *(const v8bf*)(p + 8 * half);
  v8bf hi = *(const v8bf*)(p + 16 + 8 * half);
  v16bf a;
#pragma unroll
  for (int i = 0; i < 8; ++i) { a[i] = lo[i]; a[i + 8] = hi[i]; }
  return a;
}

// B (32x16 bf16, K x N): lane L -> col N=L%16; elems 0..7 at K=8*half+i,
// elems 8..15 at K=16+8*half+i.
static __device__ inline v16bf bfrag(const __bf16* __restrict__ p, int half) {
  v8bf lo = *(const v8bf*)(p + 8 * half);
  v8bf hi = *(const v8bf*)(p + 16 + 8 * half);
  v16bf b;
#pragma unroll
  for (int i = 0; i < 8; ++i) { b[i] = lo[i]; b[i + 8] = hi[i]; }
  return b;
}

static __device__ inline v16bf zfrag() {
  v16bf z;
#pragma unroll
  for (int i = 0; i < 16; ++i) z[i] = (__bf16)0.0f;
  return z;
}

static __device__ inline v8f wmma_bf16(v16bf a, v16bf b, v8f c) {
  return __builtin_amdgcn_wmma_f32_16x16x32_bf16(false, a, false, b, (short)0,
                                                 c, false, false);
}

static __device__ inline void store_c_bf16(v8f a, __bf16* dst) {
  v8bf o;
#pragma unroll
  for (int r = 0; r < 8; ++r) o[r] = (__bf16)a[r];
  *(v8bf*)dst = o;
}

// ---------------- Prep / conversion kernels ---------------------------------

// features [B][DIN][T0] f32 -> featT [B][T0][DIN] bf16
__global__ void k_featT(const float* __restrict__ f, __bf16* __restrict__ o) {
  long i = (long)blockIdx.x * blockDim.x + threadIdx.x;  // exact grid
  int b  = (int)(i / ((long)DIN * T0_));
  long r = i % ((long)DIN * T0_);
  int ci = (int)(r / T0_);
  int t  = (int)(r % T0_);
  o[((size_t)b * T0_ + t) * DIN + ci] = (__bf16)f[i];
}

// conv w [Q][256][CIN][3] f32 -> [Q][3][256][CIN] bf16
__global__ void k_wperm(const float* __restrict__ w, __bf16* __restrict__ o,
                        int cin) {
  long i = (long)blockIdx.x * blockDim.x + threadIdx.x;  // exact grid
  int kk = (int)(i % 3);
  long r = i / 3;
  int ci = (int)(r % cin); r /= cin;
  int co = (int)(r % 256);
  int q  = (int)(r / 256);
  o[(((size_t)q * 3 + kk) * 256 + co) * cin + ci] = (__bf16)w[i];
}

// plain f32 -> bf16 (hidden_states, cls_w)
__global__ void k_cvt(const float* __restrict__ x, __bf16* __restrict__ o) {
  long i = (long)blockIdx.x * blockDim.x + threadIdx.x;  // exact grid
  o[i] = (__bf16)x[i];
}

// codebooks [Q*CBS][CBD] f32 -> L2-normalized bf16 (one wave per row)
__global__ void k_cbnorm(const float* __restrict__ cb, __bf16* __restrict__ o) {
  int lane = threadIdx.x & 31;
  int row  = blockIdx.x * 8 + (threadIdx.x >> 5);  // 8192 rows exact
  const float* r = cb + (size_t)row * CBD;
  float v[8], ss = 0.f;
#pragma unroll
  for (int j = 0; j < 8; ++j) { v[j] = r[lane + 32 * j]; ss += v[j] * v[j]; }
#pragma unroll
  for (int m = 16; m >= 1; m >>= 1) ss += __shfl_xor(ss, m, 32);
  float inv = 1.0f / fmaxf(sqrtf(ss), 1e-12f);
#pragma unroll
  for (int j = 0; j < 8; ++j)
    o[(size_t)row * CBD + lane + 32 * j] = (__bf16)(v[j] * inv);
}

// ---------------- Strided conv as 3 shifted WMMA GEMMs ----------------------
// out[(q,b)][t'][co] = sum_kk  W_kk[co][ci] * in[(q,b)][2t'+kk-1][ci]
// Each wave computes a 32(co) x 32(t') C tile with 4 accumulators.
template <int CIN, int TIN, int TOUT>
__global__ void conv_kernel(const __bf16* __restrict__ inT, size_t inQStride,
                            const __bf16* __restrict__ wT,
                            __bf16* __restrict__ outT) {
  const int lane = threadIdx.x & 31;
  const int half = lane >> 4, lm = lane & 15;
  int wid = blockIdx.x * (blockDim.x >> 5) + (threadIdx.x >> 5);
  const int NT = TOUT / 32;
  int nTile = wid % NT;  wid /= NT;
  int mTile = wid & 7;   wid >>= 3;     // 256/32 = 8 co-tiles
  int b = wid & 7, q = wid >> 3;
  int m0 = mTile * 32, n0 = nTile * 32;
  int nt0 = n0 + lm, nt1 = n0 + 16 + lm;   // the two t' columns this lane owns

  v8f acc00 = {}, acc01 = {}, acc10 = {}, acc11 = {};
  const __bf16* wr0 = wT + ((size_t)q * 3 * 256 + m0 + lm) * CIN;
  const __bf16* wr1 = wr0 + (size_t)16 * CIN;
  const __bf16* ibase = inT + (size_t)q * inQStride + (size_t)b * TIN * CIN;
#pragma unroll
  for (int kk = 0; kk < 3; ++kk) {
    int t0 = 2 * nt0 + kk - 1, t1 = 2 * nt1 + kk - 1;
    bool vld0 = (t0 >= 0) && (t0 < TIN);
    bool vld1 = (t1 >= 0) && (t1 < TIN);
    const __bf16* ar0 = wr0 + (size_t)kk * 256 * CIN;
    const __bf16* ar1 = wr1 + (size_t)kk * 256 * CIN;
    const __bf16* bc0 = ibase + (size_t)(vld0 ? t0 : 0) * CIN;
    const __bf16* bc1 = ibase + (size_t)(vld1 ? t1 : 0) * CIN;
    for (int k0 = 0; k0 < CIN; k0 += 32) {
      v16bf a0 = afrag(ar0 + k0, half);
      v16bf a1 = afrag(ar1 + k0, half);
      v16bf b0 = vld0 ? bfrag(bc0 + k0, half) : zfrag();
      v16bf b1 = vld1 ? bfrag(bc1 + k0, half) : zfrag();
      acc00 = wmma_bf16(a0, b0, acc00);
      acc01 = wmma_bf16(a0, b1, acc01);
      acc10 = wmma_bf16(a1, b0, acc10);
      acc11 = wmma_bf16(a1, b1, acc11);
    }
  }
  // transposed stores: per sub-tile, lane's 8 accum rows are contiguous co.
  __bf16* obase = outT + (size_t)(q * B_ + b) * TOUT * 256;
  store_c_bf16(acc00, obase + (size_t)(n0 + lm) * 256 + (m0 + 8 * half));
  store_c_bf16(acc01, obase + (size_t)(n0 + 16 + lm) * 256 + (m0 + 8 * half));
  store_c_bf16(acc10, obase + (size_t)(n0 + lm) * 256 + (m0 + 16 + 8 * half));
  store_c_bf16(acc11,
               obase + (size_t)(n0 + 16 + lm) * 256 + (m0 + 16 + 8 * half));
}

// ---------------- Quantize: enc . cbn^T, per-row argmax ---------------------
// One wave owns 16 rows x all 1024 codes; 16x32 tiles, 2 accumulators.
// K=256 -> the 8 A fragments (64 VGPR) stay register-resident across c-tiles.
__global__ void quant_kernel(const __bf16* __restrict__ encT,
                             const __bf16* __restrict__ cbn,
                             int* __restrict__ labels) {
  __shared__ float lds[8][16][33];
  const int lane = threadIdx.x & 31, wv = threadIdx.x >> 5;
  const int half = lane >> 4, lm = lane & 15;
  const int wid = blockIdx.x * 8 + wv;   // 4096 row-tiles exact
  const int r0 = wid * 16;
  const int q = r0 >> 13;                // /8192 (uniform in tile)
  const __bf16* ar = encT + (size_t)(r0 + lm) * CBD;
  float best = -3.0e38f; int bidx = 0;
  for (int ct = 0; ct < CBS / 32; ++ct) {
    int c0 = ct * 32;
    const __bf16* bc0 = cbn + ((size_t)q * CBS + c0 + lm) * CBD;
    const __bf16* bc1 = cbn + ((size_t)q * CBS + c0 + 16 + lm) * CBD;
    if (ct + 1 < CBS / 32)
      __builtin_prefetch(cbn + ((size_t)q * CBS + c0 + 32 + lm) * CBD, 0, 1);
    v8f acc0 = {}, acc1 = {};
#pragma unroll
    for (int k0 = 0; k0 < CBD; k0 += 32) {
      v16bf a  = afrag(ar + k0, half);     // loop-invariant across ct: hoisted
      v16bf b0 = bfrag(bc0 + k0, half);
      v16bf b1 = bfrag(bc1 + k0, half);
      acc0 = wmma_bf16(a, b0, acc0);
      acc1 = wmma_bf16(a, b1, acc1);
    }
#pragma unroll
    for (int r = 0; r < 8; ++r) {
      lds[wv][8 * half + r][lm]      = acc0[r];
      lds[wv][8 * half + r][16 + lm] = acc1[r];
    }
    __syncthreads();
    if (lane < 16) {
#pragma unroll
      for (int n = 0; n < 32; ++n) {
        float v = lds[wv][lane][n];
        if (v > best) { best = v; bidx = c0 + n; }  // strict > == first-max
      }
    }
    __syncthreads();
  }
  if (lane < 16) labels[r0 + lane] = bidx;
}

// ---------------- Fused logits + online logsumexp + NLL ---------------------
__global__ void ce_kernel(const __bf16* __restrict__ hbf,
                          const __bf16* __restrict__ cwbf,
                          const float* __restrict__ clsb,
                          const int* __restrict__ labels,
                          float* __restrict__ partial) {
  __shared__ float lds[8][16][33];
  const int lane = threadIdx.x & 31, wv = threadIdx.x >> 5;
  const int half = lane >> 4, lm = lane & 15;
  const int wid = blockIdx.x * 8 + wv;   // 4096 row-tiles exact
  const int r0 = wid * 16;
  const int q = r0 >> 13;
  const __bf16* ar = hbf + (size_t)((r0 & 8191) + lm) * HID_;
  int lbl = 0;
  if (lane < 16) lbl = labels[r0 + lane];
  float mmax = -3.0e38f, ssum = 0.f, lv = 0.f;
  for (int ct = 0; ct < CBS / 32; ++ct) {
    int c0 = ct * 32;
    float bias0 = clsb[q * CBS + c0 + lm];        // n = c0 + lm
    float bias1 = clsb[q * CBS + c0 + 16 + lm];   // n = c0 + 16 + lm
    const __bf16* bc0 = cwbf + ((size_t)q * CBS + c0 + lm) * HID_;
    const __bf16* bc1 = cwbf + ((size_t)q * CBS + c0 + 16 + lm) * HID_;
    if (ct + 1 < CBS / 32)
      __builtin_prefetch(cwbf + ((size_t)q * CBS + c0 + 32 + lm) * HID_, 0, 1);
    v8f acc0, acc1;
#pragma unroll
    for (int r = 0; r < 8; ++r) { acc0[r] = bias0; acc1[r] = bias1; }
#pragma unroll
    for (int k0 = 0; k0 < HID_; k0 += 32) {
      v16bf a  = afrag(ar + k0, half);
      v16bf b0 = bfrag(bc0 + k0, half);
      v16bf b1 = bfrag(bc1 + k0, half);
      acc0 = wmma_bf16(a, b0, acc0);
      acc1 = wmma_bf16(a, b1, acc1);
    }
#pragma unroll
    for (int r = 0; r < 8; ++r) {
      lds[wv][8 * half + r][lm]      = acc0[r];
      lds[wv][8 * half + r][16 + lm] = acc1[r];
    }
    __syncthreads();
    if (lane < 16) {
#pragma unroll
      for (int n = 0; n < 32; ++n) {
        float v = lds[wv][lane][n];
        if (v > mmax) { ssum = ssum * expf(mmax - v) + 1.f; mmax = v; }
        else          { ssum += expf(v - mmax); }
      }
      if (lbl >= c0 && lbl < c0 + 32) lv = lds[wv][lane][lbl - c0];
    }
    __syncthreads();
  }
  float nll = (lane < 16) ? ((mmax + logf(ssum)) - lv) : 0.f;
#pragma unroll
  for (int m = 16; m >= 1; m >>= 1) nll += __shfl_xor(nll, m, 32);
  if (lane == 0) partial[wid] = nll;
}

// deterministic final reduction of NPART partials
__global__ void reduce_kernel(const float* __restrict__ partial,
                              float* __restrict__ out) {
  __shared__ float s[256];
  int tid = threadIdx.x;
  float acc = 0.f;
  for (int i = tid; i < NPART; i += 256) acc += partial[i];
  s[tid] = acc;
  __syncthreads();
  for (int st = 128; st > 0; st >>= 1) {
    if (tid < st) s[tid] += s[tid + st];
    __syncthreads();
  }
  if (tid == 0) out[0] = s[0] * (1.0f / (float)NROWS);
}

// ---------------- Host launcher ---------------------------------------------
extern "C" void kernel_launch(void* const* d_in, const int* in_sizes, int n_in,
                              void* d_out, int out_size, void* d_ws,
                              size_t ws_size, hipStream_t stream) {
  (void)in_sizes; (void)n_in; (void)out_size; (void)ws_size;
  const float* features  = (const float*)d_in[0];
  const float* hidden    = (const float*)d_in[1];
  const float* conv1_w   = (const float*)d_in[2];
  const float* conv2_w   = (const float*)d_in[3];
  const float* codebooks = (const float*)d_in[4];
  const float* cls_w     = (const float*)d_in[5];
  const float* cls_b     = (const float*)d_in[6];
  float* out = (float*)d_out;

  char* W = (char*)d_ws;
  size_t off = 0;
  auto alloc = [&](size_t bytes) -> char* {
    char* p = W + off;
    off += (bytes + 255) & ~(size_t)255;
    return p;
  };
  __bf16* featT  = (__bf16*)alloc((size_t)B_ * T0_ * DIN * 2);        // 33.5 MB
  __bf16* w1T    = (__bf16*)alloc((size_t)Q_ * 3 * 256 * DIN * 2);    //  6.3 MB
  __bf16* w2T    = (__bf16*)alloc((size_t)Q_ * 3 * 256 * CBD * 2);    //  3.1 MB
  __bf16* cbn    = (__bf16*)alloc((size_t)Q_ * CBS * CBD * 2);        //  4.2 MB
  __bf16* hbf    = (__bf16*)alloc((size_t)B_ * T2_ * HID_ * 2);       // 12.6 MB
  __bf16* cwbf   = (__bf16*)alloc((size_t)Q_ * CBS * HID_ * 2);       // 12.6 MB
  __bf16* conv1T = (__bf16*)alloc((size_t)Q_ * B_ * T1_ * 256 * 2);   // 67.1 MB
  __bf16* encT   = (__bf16*)alloc((size_t)Q_ * B_ * T2_ * 256 * 2);   // 33.5 MB
  int*    labels = (int*)alloc((size_t)NROWS * 4);
  float*  partial = (float*)alloc((size_t)NPART * 4);

  // --- convert / repack (exact grids, no tail) ---
  k_featT<<<65536, 256, 0, stream>>>(features, featT);           // 16,777,216
  k_wperm<<<12288, 256, 0, stream>>>(conv1_w, w1T, DIN);         //  3,145,728
  k_wperm<<<6144, 256, 0, stream>>>(conv2_w, w2T, CBD);          //  1,572,864
  k_cbnorm<<<1024, 256, 0, stream>>>(codebooks, cbn);            //  8192 rows
  k_cvt<<<24576, 256, 0, stream>>>(hidden, hbf);                 //  6,291,456
  k_cvt<<<24576, 256, 0, stream>>>(cls_w, cwbf);                 //  6,291,456

  // --- conv1: per (q,b) GEMM [256 x 512*3] x [512*3 x 2048], 32768 waves ---
  conv_kernel<DIN, T0_, T1_><<<4096, 256, 0, stream>>>(featT, (size_t)0, w1T,
                                                       conv1T);
  // --- conv2: per (q,b) GEMM [256 x 256*3] x [256*3 x 1024], 16384 waves ---
  conv_kernel<CBD, T1_, T2_><<<2048, 256, 0, stream>>>(
      conv1T, (size_t)B_ * T1_ * 256, w2T, encT);

  // --- quantize: [65536 x 256] . [256 x 1024]^T + argmax ---
  quant_kernel<<<512, 256, 0, stream>>>(encT, cbn, labels);

  // --- fused logits + log-softmax + NLL, then deterministic mean ---
  ce_kernel<<<512, 256, 0, stream>>>(hbf, cwbf, cls_b, labels, partial);
  reduce_kernel<<<1, 256, 0, stream>>>(partial, out);
}
